// KLLoss_85315230368186
// MI455X (gfx1250) — compile-verified, matrix operand order checked by reading, not run on hardware
//
#include <hip/hip_runtime.h>
#include <math.h>

// Problem shape (from reference): B=64, K=17, H=W=128
#define B_    64
#define K_    17
#define H_    128
#define W_    128
#define N_    (H_ * W_)      // 16384 spatial elements per (b,k) row
#define ROWS  (B_ * K_)      // 1088
#define TPB   256            // 8 wave32 waves per block

typedef __attribute__((ext_vector_type(2)))  float    v2f;
typedef __attribute__((ext_vector_type(8)))  float    v8f;
typedef __attribute__((ext_vector_type(16))) _Float16 v16h;

// ---- compile-time 5x5 gaussian (sigma = 0.5), matching _gauss_kernel() ----
// g = exp(-x^2 / (2*0.25)) for x in {-2,-1,0,1,2} -> {e^-8, e^-2, 1, e^-2, e^-8}
#define GE8 3.3546262790251185e-04
#define GE2 1.3533528323661270e-01
#define GS1 (1.0 + 2.0 * GE2 + 2.0 * GE8)
#define GW(a, b) ((float)(((a) * (b)) / (GS1 * GS1)))
__device__ __constant__ float W2[25] = {
    GW(GE8,GE8), GW(GE8,GE2), GW(GE8,1.0), GW(GE8,GE2), GW(GE8,GE8),
    GW(GE2,GE8), GW(GE2,GE2), GW(GE2,1.0), GW(GE2,GE2), GW(GE2,GE8),
    GW(1.0,GE8), GW(1.0,GE2), GW(1.0,1.0), GW(1.0,GE2), GW(1.0,GE8),
    GW(GE2,GE8), GW(GE2,GE2), GW(GE2,1.0), GW(GE2,GE2), GW(GE2,GE8),
    GW(GE8,GE8), GW(GE8,GE2), GW(GE8,1.0), GW(GE8,GE2), GW(GE8,GE8),
};

// Deterministic WMMA reduction of 256 fp32 partials (in LDS) by wave 0.
// B = all-ones => D[m][n] = sum_k A[m][k]; sum of 8 D VGPRs per lane gives the
// half-matrix total, shfl_xor(16) combines the two halves.
__device__ inline float block256_wmma_sum(const float* sdata, int lane) {
#if __has_builtin(__builtin_amdgcn_wmma_f32_16x16x4_f32)
    v2f bones; bones.x = 1.0f; bones.y = 1.0f;          // 4x16 all-ones B
    v8f acc = {};                                        // C starts at 0
#pragma unroll
    for (int c = 0; c < 4; ++c) {                        // 4 x 64 partials
        v2f a;
        a.x = sdata[c * 64 + 2 * lane + 0];
        a.y = sdata[c * 64 + 2 * lane + 1];
        acc = __builtin_amdgcn_wmma_f32_16x16x4_f32(
            false, a, false, bones, (short)0, acc, false, false);
    }
#else
    // Fallback (codegen-confirmed builtin): pack 256 partials into half of a
    // 16x32 f16 A matrix (8 of 16 per-lane slots), B = ones, single WMMA.
    v16h a, bones;
#pragma unroll
    for (int i = 0; i < 16; ++i) {
        a[i]     = (i < 8) ? (_Float16)sdata[lane * 8 + i] : (_Float16)0.0f;
        bones[i] = (_Float16)1.0f;
    }
    v8f acc = {};
    acc = __builtin_amdgcn_wmma_f32_16x16x32_f16(
        false, a, false, bones, (short)0, acc, false, false);
#endif
    float s = acc[0] + acc[1] + acc[2] + acc[3] + acc[4] + acc[5] + acc[6] + acc[7];
    s += __shfl_xor(s, 16, 32);                          // combine M halves
    return s;
}

// One workgroup per (b,k) row: single HBM pass (registers cache the row),
// stable logsumexp, WMMA partial-sum reduction, analytic 25-point KL window.
__global__ __launch_bounds__(TPB) void kl_row_kernel(
    const float* __restrict__ hm,      // [ROWS, 16384]
    const float* __restrict__ tgt,     // [ROWS, 3] (x, y, vis)
    float* __restrict__ per_bk,        // [ROWS]
    float* __restrict__ visf)          // [ROWS]
{
    __shared__ float sdata[TPB];
    __shared__ float s_total;

    const int row = blockIdx.x;
    const int tid = threadIdx.x;
    const float4* src = (const float4*)(hm + (size_t)row * N_);

    // Load 64 floats/thread (16 x float4, coalesced, unit-stride across block)
    float4 v[16];
    float pmax = -3.402823466e+38f;
#pragma unroll
    for (int i = 0; i < 16; ++i) {
        v[i] = src[tid + i * TPB];
        pmax = fmaxf(pmax, fmaxf(fmaxf(v[i].x, v[i].y), fmaxf(v[i].z, v[i].w)));
    }

    // Block max (LDS tree; deterministic)
    sdata[tid] = pmax;
    __syncthreads();
    for (int s = TPB / 2; s > 0; s >>= 1) {
        if (tid < s) sdata[tid] = fmaxf(sdata[tid], sdata[tid + s]);
        __syncthreads();
    }
    const float rowmax = sdata[0];
    __syncthreads();

    // Sum of exp(x - max) from registers (no second HBM pass)
    float psum = 0.0f;
#pragma unroll
    for (int i = 0; i < 16; ++i) {
        psum += __expf(v[i].x - rowmax);
        psum += __expf(v[i].y - rowmax);
        psum += __expf(v[i].z - rowmax);
        psum += __expf(v[i].w - rowmax);
    }
    sdata[tid] = psum;
    __syncthreads();

    if (tid < 32) {                    // wave 0, EXEC all-ones for WMMA
        float total = block256_wmma_sum(sdata, tid);
        if (tid == 0) s_total = total;
    }
    __syncthreads();

    if (tid == 0) {
        const float L = __logf(s_total) + rowmax;        // logsumexp
        const float* t3 = tgt + row * 3;
        const int txi = (int)rintf(t3[0]);
        const int tyi = (int)rintf(t3[1]);
        const int vi  = (int)rintf(t3[2]);

        const float* hp = hm + (size_t)row * N_;
        float dot = 0.0f, gsum = 0.0f, glogg = 0.0f;
#pragma unroll
        for (int di = -2; di <= 2; ++di) {
            const int y = tyi + di;
            if (y < 0 || y >= H_) continue;
#pragma unroll
            for (int dj = -2; dj <= 2; ++dj) {
                const int x = txi + dj;
                if (x < 0 || x >= W_) continue;
                const float w = W2[(di + 2) * 5 + (dj + 2)];
                dot   += w * hp[y * W_ + x];
                gsum  += w;
                glogg += w * __logf(w);
            }
        }
        per_bk[row] = glogg - dot + L * gsum;
        visf[row]   = (vi > 0) ? 1.0f : 0.0f;
    }
}

// Deterministic batchmean over visible keypoints (single block, fixed order).
__global__ __launch_bounds__(TPB) void kl_finalize_kernel(
    const float* __restrict__ per_bk,
    const float* __restrict__ visf,
    float* __restrict__ out)
{
    __shared__ float snum[TPB];
    __shared__ float scnt[TPB];
    const int tid = threadIdx.x;
    float num = 0.0f, cnt = 0.0f;
    for (int r = tid; r < ROWS; r += TPB) {
        const float vv = visf[r];
        num += per_bk[r] * vv;
        cnt += vv;
    }
    snum[tid] = num;
    scnt[tid] = cnt;
    __syncthreads();
    for (int s = TPB / 2; s > 0; s >>= 1) {
        if (tid < s) {
            snum[tid] += snum[tid + s];
            scnt[tid] += scnt[tid + s];
        }
        __syncthreads();
    }
    if (tid == 0) out[0] = snum[0] / fmaxf(scnt[0], 1.0f);
}

extern "C" void kernel_launch(void* const* d_in, const int* in_sizes, int n_in,
                              void* d_out, int out_size, void* d_ws, size_t ws_size,
                              hipStream_t stream) {
    (void)in_sizes; (void)n_in; (void)out_size; (void)ws_size;
    const float* hm  = (const float*)d_in[0];   // heatmap  [64,17,128,128] f32
    const float* tgt = (const float*)d_in[1];   // targets  [64,17,3]       f32
    float* per_bk = (float*)d_ws;               // [1088]
    float* visf   = per_bk + ROWS;              // [1088]

    kl_row_kernel<<<ROWS, TPB, 0, stream>>>(hm, tgt, per_bk, visf);
    kl_finalize_kernel<<<1, TPB, 0, stream>>>(per_bk, visf, (float*)d_out);
}
